// STWithRSbySPP_DGL_231928234429
// MI455X (gfx1250) — matrix operand order repaired and checked by reading, not compiled, not run on hardware
//
#include <hip/hip_runtime.h>
#include <cstddef>

// ---------------------------------------------------------------------------
// CDNA5 (gfx1250) implementation of STWithRSbySPP_DGL forward pass.
// All GEMMs run on v_wmma_f32_16x16x32_bf16 (wave32, 16x16 tiles, f32 acc).
// LSTM steps fuse both directions per launch and stage shared A-tiles into
// LDS with global_load_async_to_lds_b128 (+ s_wait_asynccnt).
// ---------------------------------------------------------------------------

typedef __attribute__((ext_vector_type(16))) __bf16 v16bf;
typedef __attribute__((ext_vector_type(8)))  float  v8f;

struct Frag { union { v16bf v; unsigned int u[8]; __bf16 h[16]; }; };

__device__ __forceinline__ __bf16 f32_to_bf16(float f) {
  unsigned int u = __float_as_uint(f);
  unsigned int r = (u + 0x7FFFu + ((u >> 16) & 1u)) >> 16;
  unsigned short s = (unsigned short)r;
  __bf16 out;
  __builtin_memcpy(&out, &s, 2);
  return out;
}

__device__ __forceinline__ float sigm(float x) { return 1.0f / (1.0f + __expf(-x)); }

// A fragment (16x32 bf16, M x K), row-major A, base pre-offset to (m0,k0).
// Layout: lanes 0-15 row M=lane hold K=0..15 (2 per VGPR);
// lanes 16-31 same rows hold K=16..31.
__device__ __forceinline__ void load_a(Frag& a, const __bf16* base, int lda, int lane) {
  int half = lane >> 4, m = lane & 15;
  const __bf16* row = base + (size_t)m * lda;
#pragma unroll
  for (int i = 0; i < 8; ++i) {
    int kk = 16 * half + 2 * i;
    a.u[i] = *(const unsigned int*)(row + kk);
  }
}

// B fragment (32x16 bf16, K x N) sourced from an N x K row-major matrix
// (weight style: B[k][n] = W[n][k]).  Lanes 0-15 hold K=0..15 (2 per VGPR),
// lanes 16-31 hold K=16..31; lane -> column n = lane&15.
__device__ __forceinline__ void load_b(Frag& b, const __bf16* base, int ldb, int lane) {
  int half = lane >> 4, n = lane & 15;
  const __bf16* row = base + (size_t)n * ldb + 16 * half;
#pragma unroll
  for (int i = 0; i < 8; ++i) b.u[i] = *(const unsigned int*)(row + 2 * i);
}

// B fragment from a K x N row-major matrix (column gather, strided 2B loads).
__device__ __forceinline__ void load_b_col(Frag& b, const __bf16* base, int ldk, int lane) {
  int half = lane >> 4, n = lane & 15;
  const __bf16* p = base + n + (size_t)(16 * half) * ldk;
#pragma unroll
  for (int e = 0; e < 16; ++e) b.h[e] = p[(size_t)e * ldk];
}

__device__ __forceinline__ v8f wmma_bf16(const Frag& a, const Frag& b, v8f c) {
  return __builtin_amdgcn_wmma_f32_16x16x32_bf16(false, a.v, false, b.v,
                                                 (short)0, c, false, false);
}

__device__ __forceinline__ v8f vzero8() {
  v8f z = {0.f, 0.f, 0.f, 0.f, 0.f, 0.f, 0.f, 0.f};
  return z;
}

// Flat pointer into LDS -> wave-relative LDS byte offset (ISA 10.2: LDS
// aperture maps addr[31:0] to the LDS logical address).
__device__ __forceinline__ unsigned lds_off_of(const void* p) {
  return (unsigned)(unsigned long long)p;
}

// Async global -> LDS copy of one 16B chunk (ASYNCcnt-tracked, TDM-free path).
__device__ __forceinline__ void async_copy_b128(unsigned lds_off, const void* gptr) {
  unsigned long long ga = (unsigned long long)gptr;
  asm volatile("global_load_async_to_lds_b128 %0, %1, off"
               :: "v"(lds_off), "v"(ga) : "memory");
}

__device__ __forceinline__ void async_wait0() {
  asm volatile("s_wait_asynccnt 0" ::: "memory");
}

// ------------------------------ utility kernels ----------------------------

__global__ void cvt_bf16_k(const float* __restrict__ in, __bf16* __restrict__ out, int n) {
  int i = blockIdx.x * 256 + threadIdx.x;
  if (i < n) out[i] = f32_to_bf16(in[i]);
}

__global__ void add2_k(const float* a, const float* b, float* out, int n) {
  int i = blockIdx.x * 256 + threadIdx.x;
  if (i < n) out[i] = a[i] + b[i];
}

__global__ void zero_k(float* p, int n) {
  int i = blockIdx.x * 256 + threadIdx.x;
  if (i < n) p[i] = 0.f;
}

__global__ void init_state_k(const float* h0, const float* c0, __bf16* h, float* c, int n) {
  int i = blockIdx.x * 256 + threadIdx.x;
  if (i < n) { h[i] = f32_to_bf16(h0[i]); c[i] = c0[i]; }
}

// documents (N=2048, T=80, W=128) -> Xbf (T, N, W) bf16
__global__ void pack_docs_k(const float* __restrict__ doc, __bf16* __restrict__ X) {
  size_t idx = (size_t)blockIdx.x * 256 + threadIdx.x;   // exact 20971520 threads
  int w = (int)(idx & 127);
  size_t r = idx >> 7;
  size_t t = r % 80, n = r / 80;
  X[(t * 2048 + n) * 128 + w] = f32_to_bf16(doc[idx]);
}

__global__ void finish_sentpres_k(const float* hf, const float* hb, __bf16* sp) {
  int i = blockIdx.x * 256 + threadIdx.x;                // 2048*256
  int n = i >> 8, j = i & 255;
  float v = (j < 128) ? hf[n * 128 + j] : hb[n * 128 + (j - 128)];
  sp[i] = f32_to_bf16(tanhf(v * (1.0f / 80.0f)));
}

// tag hidden states (T=64, N=32, H=128) per dir -> tagout (B*64, 256) tanh'd
__global__ void tagout_k(const float* hf, const float* hb,
                         float* outF, __bf16* outB) {
  int i = blockIdx.x * 256 + threadIdx.x;                // 2048*256
  int row = i >> 8, j = i & 255;
  int b = row >> 6, d = row & 63;
  float v = (j < 128) ? tanhf(hf[(d * 32 + b) * 128 + j])
                      : tanhf(hb[(d * 32 + b) * 128 + (j - 128)]);
  outF[i] = v;
  outB[i] = f32_to_bf16(v);
}

// row-wise L2 normalize (clip 1e-8) -> bf16, one wave per row
__global__ __launch_bounds__(256) void norm_k(const float* __restrict__ H,
                                              __bf16* __restrict__ HN) {
  int lane = threadIdx.x & 31;
  int row = blockIdx.x * 8 + (threadIdx.x >> 5);
  const float* r = H + (size_t)row * 256;
  float ss = 0.f;
  for (int j = lane; j < 256; j += 32) { float v = r[j]; ss += v * v; }
#pragma unroll
  for (int off = 16; off; off >>= 1) ss += __shfl_xor(ss, off, 32);
  float inv = 1.0f / fmaxf(sqrtf(ss), 1e-8f);
  for (int j = lane; j < 256; j += 32) HN[(size_t)row * 256 + j] = f32_to_bf16(r[j] * inv);
}

// ------------------------------- LSTM step ---------------------------------
// One timestep, BOTH directions (blockIdx.y = dir). Block = 16 batch rows,
// 256 threads (8 waves). The 16xKin x-tile and 16x128 h-tile (shared by all
// 8 waves as WMMA A-operands) are staged into LDS with async global->LDS
// copies; gates are computed with WMMA, staged in LDS, then the elementwise
// gate math updates h (bf16, in place) and c (f32).
__global__ __launch_bounds__(256) void lstm_step2_k(
    const __bf16* __restrict__ xtF, const __bf16* __restrict__ xtB,
    int ldx, long dxF, long dxB,
    const __bf16* __restrict__ WihF, const __bf16* __restrict__ WihB, int Kin,
    const __bf16* __restrict__ WhhF, const __bf16* __restrict__ WhhB,
    const float* __restrict__ biasF, const float* __restrict__ biasB,
    __bf16* __restrict__ hSF, __bf16* __restrict__ hSB,
    float* __restrict__ cSF, float* __restrict__ cSB,
    float* __restrict__ hsumF, float* __restrict__ hsumB,
    float* __restrict__ hstF, float* __restrict__ hstB) {
  __shared__ float g[16 * 512];
  __shared__ __align__(16) __bf16 xa[16 * 256];
  __shared__ __align__(16) __bf16 ha[16 * 128];

  const int dir = blockIdx.y;
  const __bf16* xt   = dir ? xtB : xtF;
  const __bf16* Wih  = dir ? WihB : WihF;
  const __bf16* Whh  = dir ? WhhB : WhhF;
  const float*  bias = dir ? biasB : biasF;
  __bf16* h_state = dir ? hSB : hSF;
  float*  c_state = dir ? cSB : cSF;
  float*  hsum    = dir ? hsumB : hsumF;
  float*  hstore  = dir ? hstB : hstF;
  long    dxt     = dir ? dxB : dxF;

  int lane = threadIdx.x & 31, wave = threadIdx.x >> 5, tid = threadIdx.x;
  int m0 = blockIdx.x * 16;

  // --- async stage A-tiles into LDS (16B chunks, 1-2 per thread) ---
  {
    int cpr = Kin >> 3;                       // 16B chunks per x row
    for (int c0 = tid; c0 < 16 * cpr; c0 += 256) {
      int row = c0 / cpr, ch = c0 - row * cpr;
      async_copy_b128(lds_off_of(&xa[row * Kin + ch * 8]),
                      xt + (size_t)(m0 + row) * ldx + ch * 8);
    }
    int row = tid >> 4, ch = tid & 15;        // 16 chunks per h row, 256 total
    async_copy_b128(lds_off_of(&ha[row * 128 + ch * 8]),
                    h_state + (size_t)(m0 + row) * 128 + ch * 8);
    async_wait0();
    __syncthreads();
  }

  // prefetch next timestep's x tile (global_prefetch_b8), one line per row
  if (tid < 16)
    __builtin_prefetch((const char*)(xt + (size_t)(m0 + tid) * ldx) + dxt, 0, 3);

  v8f acc[4];
#pragma unroll
  for (int j = 0; j < 4; ++j) acc[j] = vzero8();

  for (int k0 = 0; k0 < Kin; k0 += 32) {
    Frag a; load_a(a, xa + k0, Kin, lane);                       // LDS A
#pragma unroll
    for (int j = 0; j < 4; ++j) {
      Frag b; load_b(b, Wih + (size_t)((wave * 4 + j) * 16) * Kin + k0, Kin, lane);
      acc[j] = wmma_bf16(a, b, acc[j]);
    }
  }
#pragma unroll
  for (int k0 = 0; k0 < 128; k0 += 32) {
    Frag a; load_a(a, ha + k0, 128, lane);                       // LDS A
#pragma unroll
    for (int j = 0; j < 4; ++j) {
      Frag b; load_b(b, Whh + (size_t)((wave * 4 + j) * 16) * 128 + k0, 128, lane);
      acc[j] = wmma_bf16(a, b, acc[j]);
    }
  }
  int half = lane >> 4;
#pragma unroll
  for (int j = 0; j < 4; ++j) {
    int col = (wave * 4 + j) * 16 + (lane & 15);
#pragma unroll
    for (int r = 0; r < 8; ++r) g[(r + 8 * half) * 512 + col] = acc[j][r] + bias[col];
  }
  __syncthreads();
#pragma unroll
  for (int it = 0; it < 8; ++it) {
    int e = tid + it * 256;                                      // 16*128 elems
    int row = e >> 7, jj = e & 127;
    float ig = g[row * 512 + jj];
    float fg = g[row * 512 + 128 + jj];
    float gg = g[row * 512 + 256 + jj];
    float og = g[row * 512 + 384 + jj];
    size_t idx = (size_t)(m0 + row) * 128 + jj;
    float cn = sigm(fg) * c_state[idx] + sigm(ig) * tanhf(gg);
    float hn = sigm(og) * tanhf(cn);
    c_state[idx] = cn;
    h_state[idx] = f32_to_bf16(hn);                              // block-owned rows
    if (hsum)   hsum[idx] += hn;
    if (hstore) hstore[idx] = hn;
  }
}

// ------------------------------ generic GEMM -------------------------------
// C(MxN) = act(A_bf(MxK) @ W_bf(NxK)^T + bias). grid = (M/16, N/128).
template <int ACT>
__global__ __launch_bounds__(256) void gemm_k(
    const __bf16* __restrict__ A, const __bf16* __restrict__ W,
    const float* __restrict__ bias,
    float* __restrict__ outF, __bf16* __restrict__ outB, int N, int K) {
  int lane = threadIdx.x & 31, wave = threadIdx.x >> 5;
  int m0 = blockIdx.x * 16;
  int n0 = (blockIdx.y * 8 + wave) * 16;
  v8f c = vzero8();
  for (int k0 = 0; k0 < K; k0 += 32) {
    Frag a, b;
    load_a(a, A + (size_t)m0 * K + k0, K, lane);
    load_b(b, W + (size_t)n0 * K + k0, K, lane);
    c = wmma_bf16(a, b, c);
  }
  int half = lane >> 4, col = n0 + (lane & 15);
#pragma unroll
  for (int r = 0; r < 8; ++r) {
    int row = m0 + r + 8 * half;
    float v = c[r] + bias[col];
    if (ACT == 1) v = fmaxf(v, 0.f);
    if (outF) outF[(size_t)row * N + col] = v;
    if (outB) outB[(size_t)row * N + col] = f32_to_bf16(v);
  }
}

// SPP: one block per batch b. s = tanh(K Q^T / 16) (64x64, WMMA into LDS),
// then adaptive max-pool to levels 1,2,4,8 -> 15 features per row.
__global__ __launch_bounds__(256) void spp_k(const __bf16* __restrict__ Km,
                                             const __bf16* __restrict__ Qm,
                                             float* __restrict__ outFt) {
  __shared__ float s[64 * 64];
  int b = blockIdx.x, lane = threadIdx.x & 31, wave = threadIdx.x >> 5;
  const __bf16* Kb = Km + (size_t)b * 64 * 256;
  const __bf16* Qb = Qm + (size_t)b * 64 * 256;
#pragma unroll
  for (int j = 0; j < 2; ++j) {
    int tile = wave * 2 + j;
    int m0 = (tile >> 2) * 16, n0 = (tile & 3) * 16;
    v8f c = vzero8();
    for (int k0 = 0; k0 < 256; k0 += 32) {
      Frag a, bb;
      load_a(a, Kb + (size_t)m0 * 256 + k0, 256, lane);
      load_b(bb, Qb + (size_t)n0 * 256 + k0, 256, lane);
      c = wmma_bf16(a, bb, c);
    }
    int half = lane >> 4, col = n0 + (lane & 15);
#pragma unroll
    for (int r = 0; r < 8; ++r)
      s[(m0 + r + 8 * half) * 64 + col] = tanhf(c[r] * 0.0625f);
  }
  __syncthreads();
  if (threadIdx.x < 64) {
    int i = threadIdx.x;
    float e[8];
#pragma unroll
    for (int q = 0; q < 8; ++q) {
      float m = s[i * 64 + q * 8];
#pragma unroll
      for (int t = 1; t < 8; ++t) m = fmaxf(m, s[i * 64 + q * 8 + t]);
      e[q] = m;
    }
    float q4[4]; for (int q = 0; q < 4; ++q) q4[q] = fmaxf(e[2 * q], e[2 * q + 1]);
    float h2[2]; h2[0] = fmaxf(q4[0], q4[1]); h2[1] = fmaxf(q4[2], q4[3]);
    float* o = outFt + ((size_t)b * 64 + i) * 15;
    o[0] = fmaxf(h2[0], h2[1]); o[1] = h2[0]; o[2] = h2[1];
    for (int q = 0; q < 4; ++q) o[3 + q] = q4[q];
    for (int q = 0; q < 8; ++q) o[7 + q] = e[q];
  }
}

// cosine-similarity matrix: C = hn hn^T per batch (64x64 bf16 out)
__global__ __launch_bounds__(256) void cmat_k(const __bf16* __restrict__ HN,
                                              __bf16* __restrict__ Cm) {
  int b = blockIdx.x, lane = threadIdx.x & 31, wave = threadIdx.x >> 5;
  const __bf16* Hb = HN + (size_t)b * 64 * 256;
#pragma unroll
  for (int j = 0; j < 2; ++j) {
    int tile = wave * 2 + j;
    int m0 = (tile >> 2) * 16, n0 = (tile & 3) * 16;
    v8f c = vzero8();
    for (int k0 = 0; k0 < 256; k0 += 32) {
      Frag a, bb;
      load_a(a, Hb + (size_t)m0 * 256 + k0, 256, lane);
      load_b(bb, Hb + (size_t)n0 * 256 + k0, 256, lane);
      c = wmma_bf16(a, bb, c);
    }
    int half = lane >> 4, col = n0 + (lane & 15);
#pragma unroll
    for (int r = 0; r < 8; ++r)
      Cm[((size_t)b * 64 + m0 + r + 8 * half) * 64 + col] = f32_to_bf16(c[r]);
  }
}

// neigh = (C h + h) / 65 per batch (C symmetric so C^T h == C h), bf16 out
__global__ __launch_bounds__(256) void neigh_k(const __bf16* __restrict__ Cm,
                                               const __bf16* __restrict__ Hbf,
                                               const float* __restrict__ Hf,
                                               __bf16* __restrict__ outB) {
  int b = blockIdx.x, lane = threadIdx.x & 31, wave = threadIdx.x >> 5;
#pragma unroll
  for (int j = 0; j < 8; ++j) {
    int tile = wave * 8 + j;                  // 4 x 16 tiles of the 64x256 out
    int m0 = (tile >> 4) * 16, n0 = (tile & 15) * 16;
    v8f c = vzero8();
    for (int k0 = 0; k0 < 64; k0 += 32) {
      Frag a, bb;
      load_a(a, Cm + ((size_t)b * 64 + m0) * 64 + k0, 64, lane);
      load_b_col(bb, Hbf + ((size_t)b * 64 + k0) * 256 + n0, 256, lane);
      c = wmma_bf16(a, bb, c);
    }
    int half = lane >> 4, col = n0 + (lane & 15);
#pragma unroll
    for (int r = 0; r < 8; ++r) {
      size_t idx = ((size_t)b * 64 + m0 + r + 8 * half) * 256 + col;
      outB[idx] = f32_to_bf16((c[r] + Hf[idx]) * (1.0f / 65.0f));
    }
  }
}

// tiny 8-class head + log_softmax (VALU; 2048 rows x 8x286 MACs)
__global__ void classifier_k(const float* __restrict__ dgl,
                             const float* __restrict__ sentFt,
                             const float* __restrict__ roleFt,
                             const float* __restrict__ clsW,
                             const float* __restrict__ clsb,
                             float* __restrict__ out) {
  int row = blockIdx.x * 256 + threadIdx.x;
  if (row >= 2048) return;
  float lg[8];
#pragma unroll
  for (int c = 0; c < 8; ++c) {
    const float* w = clsW + c * 286;
    float acc = clsb[c];
    for (int f = 0; f < 256; ++f) acc += dgl[(size_t)row * 256 + f] * w[f];
    for (int f = 0; f < 15; ++f)  acc += sentFt[row * 15 + f] * w[256 + f];
    for (int f = 0; f < 15; ++f)  acc += roleFt[row * 15 + f] * w[271 + f];
    lg[c] = acc;
  }
  float m = lg[0];
#pragma unroll
  for (int c = 1; c < 8; ++c) m = fmaxf(m, lg[c]);
  float s = 0.f;
#pragma unroll
  for (int c = 0; c < 8; ++c) s += __expf(lg[c] - m);
  float lse = logf(s);
#pragma unroll
  for (int c = 0; c < 8; ++c) out[(size_t)row * 8 + c] = lg[c] - m - lse;
}

// ------------------------------- host driver -------------------------------

extern "C" void kernel_launch(void* const* d_in, const int* in_sizes, int n_in,
                              void* d_out, int out_size, void* d_ws, size_t ws_size,
                              hipStream_t stream) {
  (void)in_sizes; (void)n_in; (void)out_size; (void)ws_size;
  const float* doc    = (const float*)d_in[0];
  const float* sWihF  = (const float*)d_in[2];
  const float* sWhhF  = (const float*)d_in[3];
  const float* sbihF  = (const float*)d_in[4];
  const float* sbhhF  = (const float*)d_in[5];
  const float* sWihB  = (const float*)d_in[6];
  const float* sWhhB  = (const float*)d_in[7];
  const float* sbihB  = (const float*)d_in[8];
  const float* sbhhB  = (const float*)d_in[9];
  const float* sH0    = (const float*)d_in[10];
  const float* sC0    = (const float*)d_in[11];
  const float* tWihF  = (const float*)d_in[12];
  const float* tWhhF  = (const float*)d_in[13];
  const float* tbihF  = (const float*)d_in[14];
  const float* tbhhF  = (const float*)d_in[15];
  const float* tWihB  = (const float*)d_in[16];
  const float* tWhhB  = (const float*)d_in[17];
  const float* tbihB  = (const float*)d_in[18];
  const float* tbhhB  = (const float*)d_in[19];
  const float* tH0    = (const float*)d_in[20];
  const float* tC0    = (const float*)d_in[21];
  const float* sfWK   = (const float*)d_in[22];
  const float* sfbK   = (const float*)d_in[23];
  const float* sfWQ   = (const float*)d_in[24];
  const float* sfbQ   = (const float*)d_in[25];
  const float* rfWK   = (const float*)d_in[26];
  const float* rfbK   = (const float*)d_in[27];
  const float* rfWQ   = (const float*)d_in[28];
  const float* rfbQ   = (const float*)d_in[29];
  const float* sageW  = (const float*)d_in[30];
  const float* sageB  = (const float*)d_in[31];
  const float* trW    = (const float*)d_in[32];
  const float* trB    = (const float*)d_in[33];
  const float* clsW   = (const float*)d_in[34];
  const float* clsB   = (const float*)d_in[35];

  char* wsp = (char*)d_ws;
  size_t off = 0;
  auto ALLOC = [&](size_t bytes) -> void* {
    void* p = wsp + off;
    off += (bytes + 255) & ~(size_t)255;
    return p;
  };

  __bf16* Xbf      = (__bf16*)ALLOC((size_t)20971520 * 2);
  __bf16* wihF     = (__bf16*)ALLOC(65536 * 2);
  __bf16* whhF     = (__bf16*)ALLOC(65536 * 2);
  __bf16* wihB     = (__bf16*)ALLOC(65536 * 2);
  __bf16* whhB     = (__bf16*)ALLOC(65536 * 2);
  __bf16* twihF    = (__bf16*)ALLOC(131072 * 2);
  __bf16* twhhF    = (__bf16*)ALLOC(65536 * 2);
  __bf16* twihB    = (__bf16*)ALLOC(131072 * 2);
  __bf16* twhhB    = (__bf16*)ALLOC(65536 * 2);
  __bf16* sfWKb    = (__bf16*)ALLOC(65536 * 2);
  __bf16* sfWQb    = (__bf16*)ALLOC(65536 * 2);
  __bf16* rfWKb    = (__bf16*)ALLOC(65536 * 2);
  __bf16* rfWQb    = (__bf16*)ALLOC(65536 * 2);
  __bf16* sageWb   = (__bf16*)ALLOC(65536 * 2);
  __bf16* trWb     = (__bf16*)ALLOC(65536 * 2);
  float*  biasSF   = (float*)ALLOC(512 * 4);
  float*  biasSB   = (float*)ALLOC(512 * 4);
  float*  biasTF   = (float*)ALLOC(512 * 4);
  float*  biasTB   = (float*)ALLOC(512 * 4);
  __bf16* h_sf     = (__bf16*)ALLOC(262144 * 2);
  float*  c_sf     = (float*)ALLOC(262144 * 4);
  __bf16* h_sb     = (__bf16*)ALLOC(262144 * 2);
  float*  c_sb     = (float*)ALLOC(262144 * 4);
  float*  hsumF    = (float*)ALLOC(262144 * 4);
  float*  hsumB    = (float*)ALLOC(262144 * 4);
  __bf16* sentpres = (__bf16*)ALLOC(524288 * 2);
  __bf16* Kmat     = (__bf16*)ALLOC(524288 * 2);
  __bf16* Qmat     = (__bf16*)ALLOC(524288 * 2);
  float*  sentFt   = (float*)ALLOC(30720 * 4);
  float*  roleFt   = (float*)ALLOC(30720 * 4);
  float*  tagFh    = (float*)ALLOC(262144 * 4);
  float*  tagBh    = (float*)ALLOC(262144 * 4);
  __bf16* h_tf     = (__bf16*)ALLOC(4096 * 2);
  float*  c_tf     = (float*)ALLOC(4096 * 4);
  __bf16* h_tb     = (__bf16*)ALLOC(4096 * 2);
  float*  c_tb     = (float*)ALLOC(4096 * 4);
  float*  tagoutF  = (float*)ALLOC(524288 * 4);
  __bf16* tagoutB  = (__bf16*)ALLOC(524288 * 2);
  __bf16* hnB      = (__bf16*)ALLOC(524288 * 2);
  __bf16* cmatB    = (__bf16*)ALLOC(131072 * 2);
  __bf16* neighB   = (__bf16*)ALLOC(524288 * 2);
  __bf16* h1B      = (__bf16*)ALLOC(524288 * 2);
  float*  dglF     = (float*)ALLOC(524288 * 4);

#define CVT(src, dst, n) cvt_bf16_k<<<((n) + 255) / 256, 256, 0, stream>>>((src), (dst), (n))

  // --- pack inputs / weights to bf16 ---
  pack_docs_k<<<81920, 256, 0, stream>>>(doc, Xbf);
  CVT(sWihF, wihF, 65536);  CVT(sWhhF, whhF, 65536);
  CVT(sWihB, wihB, 65536);  CVT(sWhhB, whhB, 65536);
  CVT(tWihF, twihF, 131072); CVT(tWhhF, twhhF, 65536);
  CVT(tWihB, twihB, 131072); CVT(tWhhB, twhhB, 65536);
  CVT(sfWK, sfWKb, 65536);  CVT(sfWQ, sfWQb, 65536);
  CVT(rfWK, rfWKb, 65536);  CVT(rfWQ, rfWQb, 65536);
  CVT(sageW, sageWb, 65536); CVT(trW, trWb, 65536);
  add2_k<<<2, 256, 0, stream>>>(sbihF, sbhhF, biasSF, 512);
  add2_k<<<2, 256, 0, stream>>>(sbihB, sbhhB, biasSB, 512);
  add2_k<<<2, 256, 0, stream>>>(tbihF, tbhhF, biasTF, 512);
  add2_k<<<2, 256, 0, stream>>>(tbihB, tbhhB, biasTB, 512);

  // --- init recurrent state ---
  init_state_k<<<1024, 256, 0, stream>>>(sH0, sC0, h_sf, c_sf, 262144);
  init_state_k<<<1024, 256, 0, stream>>>(sH0 + 262144, sC0 + 262144, h_sb, c_sb, 262144);
  init_state_k<<<16, 256, 0, stream>>>(tH0, tC0, h_tf, c_tf, 4096);
  init_state_k<<<16, 256, 0, stream>>>(tH0 + 4096, tC0 + 4096, h_tb, c_tb, 4096);
  zero_k<<<1024, 256, 0, stream>>>(hsumF, 262144);
  zero_k<<<1024, 256, 0, stream>>>(hsumB, 262144);

  // --- sentence BiLSTM (N=2048, T=80, K_in=128), fwd+bwd fused per step ---
  const long sStep = (long)2048 * 128 * 2;   // bytes per timestep
  for (int t = 0; t < 80; ++t) {
    long dF = (t < 79) ? sStep : 0, dB = (t < 79) ? -sStep : 0;
    lstm_step2_k<<<dim3(128, 2), 256, 0, stream>>>(
        Xbf + (size_t)t * 2048 * 128, Xbf + (size_t)(79 - t) * 2048 * 128,
        128, dF, dB, wihF, wihB, 128, whhF, whhB, biasSF, biasSB,
        h_sf, h_sb, c_sf, c_sb, hsumF, hsumB, nullptr, nullptr);
  }
  finish_sentpres_k<<<2048, 256, 0, stream>>>(hsumF, hsumB, sentpres);

  // --- sentence SPP features ---
  gemm_k<0><<<dim3(128, 2), 256, 0, stream>>>(sentpres, sfWKb, sfbK, nullptr, Kmat, 256, 256);
  gemm_k<0><<<dim3(128, 2), 256, 0, stream>>>(sentpres, sfWQb, sfbQ, nullptr, Qmat, 256, 256);
  spp_k<<<32, 256, 0, stream>>>(Kmat, Qmat, sentFt);

  // --- tag BiLSTM (N=32, T=64, K_in=256), rows stride 64*256, fused dirs ---
  const long tStep = (long)256 * 2;          // bytes per timestep
  for (int s = 0; s < 64; ++s) {
    long dF = (s < 63) ? tStep : 0, dB = (s < 63) ? -tStep : 0;
    lstm_step2_k<<<dim3(2, 2), 256, 0, stream>>>(
        sentpres + (size_t)s * 256, sentpres + (size_t)(63 - s) * 256,
        64 * 256, dF, dB, twihF, twihB, 256, twhhF, twhhB, biasTF, biasTB,
        h_tf, h_tb, c_tf, c_tb, nullptr, nullptr,
        tagFh + (size_t)s * 32 * 128, tagBh + (size_t)(63 - s) * 32 * 128);
  }
  tagout_k<<<2048, 256, 0, stream>>>(tagFh, tagBh, tagoutF, tagoutB);

  // --- role SPP features ---
  gemm_k<0><<<dim3(128, 2), 256, 0, stream>>>(tagoutB, rfWKb, rfbK, nullptr, Kmat, 256, 256);
  gemm_k<0><<<dim3(128, 2), 256, 0, stream>>>(tagoutB, rfWQb, rfbQ, nullptr, Qmat, 256, 256);
  spp_k<<<32, 256, 0, stream>>>(Kmat, Qmat, roleFt);

  // --- graph branch: cosine sim, SAGE 'gcn', transition ---
  norm_k<<<256, 256, 0, stream>>>(tagoutF, hnB);
  cmat_k<<<32, 256, 0, stream>>>(hnB, cmatB);
  neigh_k<<<32, 256, 0, stream>>>(cmatB, tagoutB, tagoutF, neighB);
  gemm_k<1><<<dim3(128, 2), 256, 0, stream>>>(neighB, sageWb, sageB, nullptr, h1B, 256, 256);
  gemm_k<0><<<dim3(128, 2), 256, 0, stream>>>(h1B, trWb, trB, dglF, nullptr, 256, 256);

  // --- classifier + log_softmax ---
  classifier_k<<<8, 256, 0, stream>>>(dglF, sentFt, roleFt, clsW, clsB, (float*)d_out);

#undef CVT
}
